// TemporalGNN_53446573031860
// MI455X (gfx1250) — matrix operand order repaired and checked by reading, not compiled, not run on hardware
//
#include <hip/hip_runtime.h>
#include <hip/hip_bf16.h>
#include <math.h>

// ---------------------------------------------------------------------------
// TemporalGNN on gfx1250: GCNConv x2 + MLP head.
// GEMMs via v_wmma_f32_16x16x32_bf16 (bf16 in, f32 accumulate),
// edge aggregation via global f32 atomics, weights staged in LDS.
// ---------------------------------------------------------------------------

typedef __attribute__((ext_vector_type(16))) __bf16 v16bf;
typedef __attribute__((ext_vector_type(8)))  __bf16 v8bf;
typedef __attribute__((ext_vector_type(8)))  float  v8f;

#define TGNN_HID 128
#define TGNN_OUTC 10

// ---------------------------------------------------------------------------
// GEMM: out[M x 128] = act(preact(A)[M x K] @ W[K x 128] (+obias)) with
//   preact(A) = relu(A + abias[k % abias_mod]) when abias != nullptr
//   act = relu when relu_out != 0
// Block: 256 threads = 8 waves; each wave owns a 16-col tile; 8 row-chunks
// of 16 rows per block. Weights transposed into LDS (bf16), A chunk staged
// per row-chunk. One v_wmma per 32-wide K chunk, accumulator chained.
// ---------------------------------------------------------------------------
__global__ __launch_bounds__(256) void tgnn_gemm128_wmma(
    const float* __restrict__ A, const float* __restrict__ W,
    const float* __restrict__ abias, int abias_mod,
    const float* __restrict__ obias, int relu_out,
    float* __restrict__ out, int M, int K)
{
    extern __shared__ char smem[];
    __bf16* Wt = (__bf16*)smem;        // [128][K]  (col-major weights: n,k)
    __bf16* At = Wt + 128 * K;         // [16][K]   (row chunk of A, bf16)

    const int tid  = threadIdx.x;
    const int wave = tid >> 5;         // 0..7 -> column tile
    const int lane = tid & 31;
    const int half = lane >> 4;        // 0 or 1
    const int l15  = lane & 15;

    // Stage W transposed into LDS as bf16 (reused by all row chunks).
    for (int i = tid; i < 128 * K; i += 256) {
        int n = i / K, k = i - n * K;
        Wt[i] = (__bf16)W[(size_t)k * 128 + n];
    }
    __syncthreads();

    for (int rc = 0; rc < 8; ++rc) {
        int row0 = (blockIdx.x * 8 + rc) * 16;
        if (row0 >= M) break;

        // Stage 16 rows of A (with optional fused pre-bias + ReLU).
        for (int i = tid; i < 16 * K; i += 256) {
            int r = i / K, k = i - r * K;
            float v = A[(size_t)(row0 + r) * K + k];
            if (abias) { v += abias[k % abias_mod]; v = fmaxf(v, 0.0f); }
            At[i] = (__bf16)v;
        }
        __syncthreads();

        v8f acc = {};
        for (int kk = 0; kk < K; kk += 32) {
            // A fragment (16x32 bf16): lane half selects K 0..7/16..23 vs 8..15/24..31
            const v8bf a0 = *(const v8bf*)&At[l15 * K + kk + half * 8];
            const v8bf a1 = *(const v8bf*)&At[l15 * K + kk + 16 + half * 8];
            // B fragment (32x16 bf16): lane = col, half selects K 0..7/16..23 vs 8..15/24..31
            const v8bf b0 = *(const v8bf*)&Wt[(wave * 16 + l15) * K + kk + half * 8];
            const v8bf b1 = *(const v8bf*)&Wt[(wave * 16 + l15) * K + kk + 16 + half * 8];
            v16bf a, b;
            #pragma unroll
            for (int j = 0; j < 8; ++j) {
                a[j] = a0[j]; a[j + 8] = a1[j];
                b[j] = b0[j]; b[j + 8] = b1[j];
            }
            acc = __builtin_amdgcn_wmma_f32_16x16x32_bf16(
                /*neg_a=*/false, a, /*neg_b=*/false, b,
                /*c_mod=*/(short)0, acc, /*reuse_a=*/false, /*reuse_b=*/false);
        }

        // C/D layout: VGPR r -> row = r + 8*half, col = lane&15 (+ tile base).
        #pragma unroll
        for (int r = 0; r < 8; ++r) {
            int mrow = half * 8 + r;
            int col  = wave * 16 + l15;
            float v = acc[r];
            if (obias)    v += obias[col];
            if (relu_out) v = fmaxf(v, 0.0f);
            out[(size_t)(row0 + mrow) * 128 + col] = v;
        }
        __syncthreads();
    }
}

// ---------------------------------------------------------------------------
// Degree / normalization kernels
// ---------------------------------------------------------------------------
__global__ void tgnn_deg_init(float* __restrict__ deg, int n)
{
    int i = blockIdx.x * blockDim.x + threadIdx.x;
    if (i < n) deg[i] = 1.0f;   // self-loop
}

__global__ void tgnn_deg_count(const int* __restrict__ col,
                               float* __restrict__ deg, int E)
{
    int e = blockIdx.x * blockDim.x + threadIdx.x;
    if (e < E) atomicAdd(&deg[col[e]], 1.0f);
}

__global__ void tgnn_deg_to_dis(float* __restrict__ deg, int n)
{
    int i = blockIdx.x * blockDim.x + threadIdx.x;
    if (i < n) deg[i] = rsqrtf(fmaxf(deg[i], 1e-12f));
}

// ---------------------------------------------------------------------------
// Aggregation: out[i] = dis[i]^2 * xw[i]  (self loop; also initializes out)
// then per-edge: out[col] += dis[row]*dis[col] * xw[row]
// ---------------------------------------------------------------------------
__global__ void tgnn_agg_self(const float* __restrict__ xw,
                              const float* __restrict__ dis,
                              float* __restrict__ out, int n)
{
    int t = blockIdx.x * blockDim.x + threadIdx.x;
    if (t >= n * TGNN_HID) return;
    int i = t >> 7;                 // / 128
    float d = dis[i];
    out[t] = d * d * xw[t];
}

__global__ void tgnn_agg_edges(const int* __restrict__ rowi,
                               const int* __restrict__ coli,
                               const float* __restrict__ dis,
                               const float* __restrict__ xw,
                               float* __restrict__ out, int E)
{
    int t = blockIdx.x * blockDim.x + threadIdx.x;   // 32 lanes per edge
    if (t >= E * 32) return;
    int e  = t >> 5;
    int c0 = (t & 31) * 4;                           // 4 channels / lane
    int r  = rowi[e];
    int c  = coli[e];
    float s = dis[r] * dis[c];
    const float4 v = *(const float4*)&xw[(size_t)r * TGNN_HID + c0];
    float* o = &out[(size_t)c * TGNN_HID + c0];
    atomicAdd(o + 0, s * v.x);
    atomicAdd(o + 1, s * v.y);
    atomicAdd(o + 2, s * v.z);
    atomicAdd(o + 3, s * v.w);
}

// ---------------------------------------------------------------------------
// Head: out[r] = log_softmax(z[r] @ Wc2 + bc2), one thread per row.
// ---------------------------------------------------------------------------
__global__ void tgnn_head(const float* __restrict__ z,
                          const float* __restrict__ Wc2,
                          const float* __restrict__ bc2,
                          float* __restrict__ out, int B)
{
    int r = blockIdx.x * blockDim.x + threadIdx.x;
    if (r >= B) return;
    float acc[TGNN_OUTC];
    #pragma unroll
    for (int j = 0; j < TGNN_OUTC; ++j) acc[j] = bc2[j];
    const float* zr = &z[(size_t)r * TGNN_HID];
    for (int k = 0; k < TGNN_HID; ++k) {
        float zv = zr[k];
        #pragma unroll
        for (int j = 0; j < TGNN_OUTC; ++j)
            acc[j] = fmaf(zv, Wc2[k * TGNN_OUTC + j], acc[j]);
    }
    float m = acc[0];
    #pragma unroll
    for (int j = 1; j < TGNN_OUTC; ++j) m = fmaxf(m, acc[j]);
    float s = 0.0f;
    #pragma unroll
    for (int j = 0; j < TGNN_OUTC; ++j) s += __expf(acc[j] - m);
    float lse = m + __logf(s);
    #pragma unroll
    for (int j = 0; j < TGNN_OUTC; ++j)
        out[(size_t)r * TGNN_OUTC + j] = acc[j] - lse;
}

// ---------------------------------------------------------------------------
// Launch
// ---------------------------------------------------------------------------
extern "C" void kernel_launch(void* const* d_in, const int* in_sizes, int n_in,
                              void* d_out, int out_size, void* d_ws, size_t ws_size,
                              hipStream_t stream) {
    (void)n_in; (void)out_size; (void)ws_size;
    const float* x   = (const float*)d_in[0];
    const int*   ei  = (const int*)d_in[1];     // [2][E]: row, col
    const float* W1  = (const float*)d_in[2];
    const float* b1  = (const float*)d_in[3];
    const float* W2  = (const float*)d_in[4];
    const float* b2  = (const float*)d_in[5];
    const float* Wc1 = (const float*)d_in[6];
    const float* bc1 = (const float*)d_in[7];
    const float* Wc2 = (const float*)d_in[8];
    const float* bc2 = (const float*)d_in[9];

    const int INC = 64;
    const int N = in_sizes[0] / INC;            // 150000
    const int E = in_sizes[1] / 2;              // 600000
    const int B = N / 3;                        // 50000 graphs

    const int* rowi = ei;
    const int* coli = ei + E;

    float* buf0 = (float*)d_ws;                       // [N x 128]
    float* buf1 = buf0 + (size_t)N * TGNN_HID;        // [N x 128]
    float* deg  = buf1 + (size_t)N * TGNN_HID;        // [N] -> becomes dis

    const int T = 256;
    auto cdiv = [](long long a, long long b) { return (int)((a + b - 1) / b); };

    // Normalization: deg = 1 + count(col); dis = rsqrt(deg)
    tgnn_deg_init <<<cdiv(N, T), T, 0, stream>>>(deg, N);
    tgnn_deg_count<<<cdiv(E, T), T, 0, stream>>>(coli, deg, E);
    tgnn_deg_to_dis<<<cdiv(N, T), T, 0, stream>>>(deg, N);

    // Layer 1: xw1 = x @ W1  (no bias here; b1 fused into layer-2 A-read)
    tgnn_gemm128_wmma<<<cdiv(N, 128), T, 288 * INC, stream>>>(
        x, W1, nullptr, 1, nullptr, 0, buf0, N, INC);
    tgnn_agg_self <<<cdiv((long long)N * TGNN_HID, T), T, 0, stream>>>(buf0, deg, buf1, N);
    tgnn_agg_edges<<<cdiv((long long)E * 32, T), T, 0, stream>>>(rowi, coli, deg, buf0, buf1, E);

    // Layer 2: xw2 = relu(agg1 + b1) @ W2
    tgnn_gemm128_wmma<<<cdiv(N, 128), T, 288 * TGNN_HID, stream>>>(
        buf1, W2, b1, TGNN_HID, nullptr, 0, buf0, N, TGNN_HID);
    tgnn_agg_self <<<cdiv((long long)N * TGNN_HID, T), T, 0, stream>>>(buf0, deg, buf1, N);
    tgnn_agg_edges<<<cdiv((long long)E * 32, T), T, 0, stream>>>(rowi, coli, deg, buf0, buf1, E);

    // Classifier hidden: z = relu( relu(agg2 + b2).reshape(B,384) @ Wc1 + bc1 )
    tgnn_gemm128_wmma<<<cdiv(B, 128), T, 288 * (3 * TGNN_HID), stream>>>(
        buf1, Wc1, b2, TGNN_HID, bc1, 1, buf0, B, 3 * TGNN_HID);

    // Output head + log_softmax
    tgnn_head<<<cdiv(B, T), T, 0, stream>>>(buf0, Wc2, bc2, (float*)d_out, B);
}
